// L2DistAttention_8022998909474
// MI455X (gfx1250) — matrix operand order repaired and checked by reading, not compiled, not run on hardware
//
#include <hip/hip_runtime.h>
#include <hip/hip_bf16.h>
#include <math.h>

// ---------------------------------------------------------------------------
// MI455X (gfx1250) degree-0 L2-distance attention block.
// Dominant compute: kv_pre = P @ W3'  (8192 x 4096 x 256, 17.2 GFLOP) where
// P[e, r*64+d] = h2[e,r]*xe[e,d] is built on the fly in registers (537 MB of
// R is never materialized), executed with v_wmma_f32_16x16x32_bf16.
// Each wave owns 4 adjacent 16x16 C tiles: one A-tile build feeds 4 WMMAs
// (4 independent accumulation chains hide XDL latency).
// ---------------------------------------------------------------------------

typedef __attribute__((ext_vector_type(16))) __bf16 bf16x16;
typedef __attribute__((ext_vector_type(8)))  __bf16 bf16x8;
typedef __attribute__((ext_vector_type(8)))  float  f32x8;
typedef __attribute__((ext_vector_type(8)))  float  v8f;

#define NODES 512     // b*n = 2*256
#define EDGES 8192    // NODES * k(16)
#define D     64
#define HID   128
#define KVD   256
#define RH    64

// ---------------- weight preprocessing ----------------
__global__ void k_convert_bf16(const float* __restrict__ in,
                               __bf16* __restrict__ out, int count) {
  int i = blockIdx.x * blockDim.x + threadIdx.x;
  if (i < count) out[i] = (__bf16)in[i];
}

// in: K x N row-major f32 -> out: N x K row-major bf16 (transposed)
__global__ void k_convert_t_bf16(const float* __restrict__ in,
                                 __bf16* __restrict__ out, int K, int N) {
  int i = blockIdx.x * blockDim.x + threadIdx.x;
  if (i >= K * N) return;
  int n = i / K, k = i % K;
  out[i] = (__bf16)in[k * N + n];
}

// ---------------- node prenorm (RMS over d, wave32 per node) ----------------
__global__ void k_node_prenorm(const float* __restrict__ features,
                               const float* __restrict__ norm_scale,
                               float* __restrict__ x) {
  int node = blockIdx.x;
  int lane = threadIdx.x;            // two channels per lane
  float f0 = features[node * D + lane];
  float f1 = features[node * D + lane + 32];
  float s = f0 * f0 + f1 * f1;
#pragma unroll
  for (int o = 16; o; o >>= 1) s += __shfl_xor(s, o, 32);
  float rms = sqrtf(s) * 0.125f;     // * d^-0.5, d = 64
  float inv = 1.0f / fmaxf(rms, 1e-12f);
  x[node * D + lane]      = f0 * inv * norm_scale[lane];
  x[node * D + lane + 32] = f1 * inv * norm_scale[lane + 32];
}

// ------- edge prep: gather xj[idx]+xi -> xe(bf16); radial layer1 -> h1 -------
__global__ void k_edge_prep(const float* __restrict__ xi,
                            const float* __restrict__ xj,
                            const int* __restrict__ nidx,
                            const float* __restrict__ rel,
                            const float* __restrict__ W1,
                            const float* __restrict__ b1,
                            const float* __restrict__ g1,
                            __bf16* __restrict__ xebf,
                            float* __restrict__ h1) {
  int e = blockIdx.x;                // edge id
  int lane = threadIdx.x;
  int bn = e >> 4;                   // node id (b*256+n)
  int b  = bn >> 8;
  int src = b * 256 + nidx[e];
  float v0 = xj[src * D + lane]      + xi[bn * D + lane];
  float v1 = xj[src * D + lane + 32] + xi[bn * D + lane + 32];
  xebf[e * D + lane]      = (__bf16)v0;
  xebf[e * D + lane + 32] = (__bf16)v1;

  float dist = rel[e];
  float a0 = dist * W1[lane]      + b1[lane];
  float a1 = dist * W1[lane + 32] + b1[lane + 32];
  a0 = a0 / (1.0f + expf(-a0));      // SiLU
  a1 = a1 / (1.0f + expf(-a1));
  float s = a0 + a1;
#pragma unroll
  for (int o = 16; o; o >>= 1) s += __shfl_xor(s, o, 32);
  float mu = s * (1.0f / 64.0f);
  float d0 = a0 - mu, d1 = a1 - mu;
  float v = d0 * d0 + d1 * d1;
#pragma unroll
  for (int o = 16; o; o >>= 1) v += __shfl_xor(v, o, 32);
  float inv = rsqrtf(v * (1.0f / 64.0f) + 1e-5f);
  h1[e * RH + lane]      = d0 * inv * g1[lane];
  h1[e * RH + lane + 32] = d1 * inv * g1[lane + 32];
}

// ------- radial layer2 activation: SiLU + LayerNorm -> h2 (bf16) -------
__global__ void k_edge_act(const float* __restrict__ h2pre,
                           const float* __restrict__ b2,
                           const float* __restrict__ g2,
                           __bf16* __restrict__ h2bf) {
  int e = blockIdx.x;
  int lane = threadIdx.x;
  float a0 = h2pre[e * RH + lane]      + b2[lane];
  float a1 = h2pre[e * RH + lane + 32] + b2[lane + 32];
  a0 = a0 / (1.0f + expf(-a0));
  a1 = a1 / (1.0f + expf(-a1));
  float s = a0 + a1;
#pragma unroll
  for (int o = 16; o; o >>= 1) s += __shfl_xor(s, o, 32);
  float mu = s * (1.0f / 64.0f);
  float d0 = a0 - mu, d1 = a1 - mu;
  float v = d0 * d0 + d1 * d1;
#pragma unroll
  for (int o = 16; o; o >>= 1) v += __shfl_xor(v, o, 32);
  float inv = rsqrtf(v * (1.0f / 64.0f) + 1e-5f);
  h2bf[e * RH + lane]      = (__bf16)(d0 * inv * g2[lane]);
  h2bf[e * RH + lane + 32] = (__bf16)(d1 * inv * g2[lane + 32]);
}

// -------- WMMA GEMM, 4 N-tiles per wave: C(MxN,f32) = A(MxK,f32)*Bt(NxK,bf16)
// A-tile layout (16-bit A 16x32): lane<16: M=lane, elems = K0..15 ;
// lane>=16: M=lane-16, K16..31.
// B-tile: lane<16: N=lane, elems = K0..15 ; lane>=16: N=lane-16, K16..31.
// One A build (f32 loads + cvt) feeds 4 independent WMMA chains.
__global__ void k_gemm_bf16_x4(const float* __restrict__ A,
                               const __bf16* __restrict__ Bt,
                               float* __restrict__ C, int N, int K) {
  int ng   = N >> 6;                       // groups of 64 columns
  int m0   = (blockIdx.x / ng) << 4;
  int n0   = (blockIdx.x % ng) << 6;
  int lane = threadIdx.x & 31;
  int hv   = lane >> 4;
  int l16  = lane & 15;
  int m = m0 + l16;
  v8f acc0 = {}, acc1 = {}, acc2 = {}, acc3 = {};
  for (int k0 = 0; k0 < K; k0 += 32) {
    const float* ap = A + (size_t)m * K + k0 + hv * 16;
    f32x8 alo = *(const f32x8*)(ap);
    f32x8 ahi = *(const f32x8*)(ap + 8);
    bf16x16 av;
#pragma unroll
    for (int i = 0; i < 8; ++i) { av[i] = (__bf16)alo[i]; av[8 + i] = (__bf16)ahi[i]; }
    const __bf16* bp = Bt + (size_t)(n0 + l16) * K + k0 + hv * 16;
    bf16x16 b0 = *(const bf16x16*)(bp);
    bf16x16 b1 = *(const bf16x16*)(bp + (size_t)16 * K);
    bf16x16 b2 = *(const bf16x16*)(bp + (size_t)32 * K);
    bf16x16 b3 = *(const bf16x16*)(bp + (size_t)48 * K);
    acc0 = __builtin_amdgcn_wmma_f32_16x16x32_bf16(false, av, false, b0, (short)0, acc0, false, false);
    acc1 = __builtin_amdgcn_wmma_f32_16x16x32_bf16(false, av, false, b1, (short)0, acc1, false, false);
    acc2 = __builtin_amdgcn_wmma_f32_16x16x32_bf16(false, av, false, b2, (short)0, acc2, false, false);
    acc3 = __builtin_amdgcn_wmma_f32_16x16x32_bf16(false, av, false, b3, (short)0, acc3, false, false);
  }
  float* cp = C + (size_t)(m0 + hv * 8) * N + n0 + l16;
#pragma unroll
  for (int v = 0; v < 8; ++v) {
    cp[(size_t)v * N]      = acc0[v];
    cp[(size_t)v * N + 16] = acc1[v];
    cp[(size_t)v * N + 32] = acc2[v];
    cp[(size_t)v * N + 48] = acc3[v];
  }
}

// ------- bilinear kernel GEMM: kv_pre[e,o] = sum_{r,dd} h2[e,r]*xe[e,dd]*
//         W3[r, o*64+dd]  +  sum_dd b3[o*64+dd]*xe[e,dd]
// K index kk = r*64+dd; each 32-chunk lies inside one r (dd0 in {0,32}), so
// B rows are 16 contiguous bf16 inside the W3bf row (no gathers).
// Block = 4 waves covering one edge-tile x all 256 output cols; xe tile is
// hoisted into 32 f32 registers/lane, h2 fetched 8-at-a-time.
__global__ void k_kv_gemm(const __bf16* __restrict__ h2bf,
                          const __bf16* __restrict__ xebf,
                          const __bf16* __restrict__ W3bf,
                          const float* __restrict__ b3,
                          float* __restrict__ kvpre) {
  int m0   = blockIdx.x << 4;          // edge-tile base
  int w    = threadIdx.x >> 5;         // wave 0..3 -> o columns w*64..w*64+63
  int lane = threadIdx.x & 31;
  int hv   = lane >> 4;
  int l16  = lane & 15;
  int e    = m0 + l16;
  int ob   = w << 6;

  // hoist xe[e, :] pieces this lane needs: chunk(c,p) = dd c*32 + hv*16 + p*8 + i
  float xef[4][8];
#pragma unroll
  for (int c = 0; c < 2; ++c)
#pragma unroll
    for (int p = 0; p < 2; ++p) {
      const __bf16* xp = xebf + (size_t)e * D + c * 32 + hv * 16 + p * 8;
#pragma unroll
      for (int i = 0; i < 8; ++i) xef[c * 2 + p][i] = (float)xp[i];
    }

  v8f acc0 = {}, acc1 = {}, acc2 = {}, acc3 = {};
  for (int r8 = 0; r8 < RH; r8 += 8) {
    bf16x8 hvec = *(const bf16x8*)(h2bf + (size_t)e * RH + r8);
#pragma unroll
    for (int rr = 0; rr < 8; ++rr) {
      int r = r8 + rr;
      float h = (float)hvec[rr];
      const __bf16* wrow = W3bf + (size_t)r * (KVD * D);
      if (r + 1 < RH)
        __builtin_prefetch(wrow + (size_t)(KVD * D) + (ob + l16) * D, 0, 1);
#pragma unroll
      for (int c = 0; c < 2; ++c) {
        bf16x16 av;
#pragma unroll
        for (int i = 0; i < 8; ++i) {
          av[i]     = (__bf16)(h * xef[c * 2][i]);
          av[8 + i] = (__bf16)(h * xef[c * 2 + 1][i]);
        }
        const __bf16* bp = wrow + (size_t)(ob + l16) * D + c * 32 + hv * 16;
        bf16x16 b0 = *(const bf16x16*)(bp);
        bf16x16 b1 = *(const bf16x16*)(bp + 16 * D);
        bf16x16 b2 = *(const bf16x16*)(bp + 32 * D);
        bf16x16 b3v = *(const bf16x16*)(bp + 48 * D);
        acc0 = __builtin_amdgcn_wmma_f32_16x16x32_bf16(false, av, false, b0,  (short)0, acc0, false, false);
        acc1 = __builtin_amdgcn_wmma_f32_16x16x32_bf16(false, av, false, b1,  (short)0, acc1, false, false);
        acc2 = __builtin_amdgcn_wmma_f32_16x16x32_bf16(false, av, false, b2,  (short)0, acc2, false, false);
        acc3 = __builtin_amdgcn_wmma_f32_16x16x32_bf16(false, av, false, b3v, (short)0, acc3, false, false);
      }
    }
  }
  // bias pass: A = xe (K = 64), B[j,o] = b3[o*64 + dd]
#pragma unroll
  for (int c = 0; c < 2; ++c) {
    bf16x16 av;
#pragma unroll
    for (int i = 0; i < 8; ++i) {
      av[i]     = (__bf16)xef[c * 2][i];
      av[8 + i] = (__bf16)xef[c * 2 + 1][i];
    }
    const float* bpf = b3 + (size_t)(ob + l16) * D + c * 32 + hv * 16;
    bf16x16 b0, b1, b2, b3v;
#pragma unroll
    for (int i = 0; i < 16; ++i) {
      b0[i]  = (__bf16)bpf[i];
      b1[i]  = (__bf16)bpf[16 * D + i];
      b2[i]  = (__bf16)bpf[32 * D + i];
      b3v[i] = (__bf16)bpf[48 * D + i];
    }
    acc0 = __builtin_amdgcn_wmma_f32_16x16x32_bf16(false, av, false, b0,  (short)0, acc0, false, false);
    acc1 = __builtin_amdgcn_wmma_f32_16x16x32_bf16(false, av, false, b1,  (short)0, acc1, false, false);
    acc2 = __builtin_amdgcn_wmma_f32_16x16x32_bf16(false, av, false, b2,  (short)0, acc2, false, false);
    acc3 = __builtin_amdgcn_wmma_f32_16x16x32_bf16(false, av, false, b3v, (short)0, acc3, false, false);
  }
  float* cp = kvpre + (size_t)(m0 + hv * 8) * KVD + ob + l16;
#pragma unroll
  for (int v = 0; v < 8; ++v) {
    cp[(size_t)v * KVD]      = acc0[v];
    cp[(size_t)v * KVD + 16] = acc1[v];
    cp[(size_t)v * KVD + 32] = acc2[v];
    cp[(size_t)v * KVD + 48] = acc3[v];
  }
}

// ---------------- L2-distance attention, wave32 per (node, head) ------------
__global__ void k_attn(const float* __restrict__ q,
                       const float* __restrict__ kv,
                       const unsigned char* __restrict__ mask,
                       float* __restrict__ outa) {
  int id = blockIdx.x;               // bn*4 + head
  int bn = id >> 2;
  int h  = id & 3;
  int lane = threadIdx.x;            // dh index 0..31
  const float NEGMAX = -3.402823466e38f;
  const float scale  = 0.17677669529663689f;   // 32^-0.5
  float qv = q[(size_t)bn * HID + h * 32 + lane];
  float sim[16];
#pragma unroll
  for (int j = 0; j < 16; ++j) {
    int e = bn * 16 + j;
    float kvv = kv[(size_t)e * KVD + h * 32 + lane];
    float d0 = qv - kvv + 1e-6f;
    float s = d0 * d0;
#pragma unroll
    for (int o = 16; o; o >>= 1) s += __shfl_xor(s, o, 32);
    float sj = -sqrtf(s) * scale;
    sim[j] = mask[e] ? sj : NEGMAX;
  }
  float mx = NEGMAX;
#pragma unroll
  for (int j = 0; j < 16; ++j) mx = fmaxf(mx, sim[j]);
  float wgt[16];
  float denom = 0.0f;
#pragma unroll
  for (int j = 0; j < 16; ++j) { wgt[j] = expf(sim[j] - mx); denom += wgt[j]; }
  float accv = 0.0f;
#pragma unroll
  for (int j = 0; j < 16; ++j)
    accv += wgt[j] * kv[(size_t)(bn * 16 + j) * KVD + HID + h * 32 + lane];
  outa[(size_t)bn * HID + h * 32 + lane] = accv / denom;
}

// ---------------------------------------------------------------------------
extern "C" void kernel_launch(void* const* d_in, const int* in_sizes, int n_in,
                              void* d_out, int out_size, void* d_ws, size_t ws_size,
                              hipStream_t stream) {
  (void)in_sizes; (void)n_in; (void)out_size; (void)ws_size;
  const float* features   = (const float*)d_in[0];
  const int*   nidx       = (const int*)  d_in[1];
  const unsigned char* nmask = (const unsigned char*)d_in[2];
  const float* rel_dist   = (const float*)d_in[3];
  const float* norm_scale = (const float*)d_in[4];
  const float* Wq   = (const float*)d_in[5];
  const float* Wxi  = (const float*)d_in[6];
  const float* Wxj  = (const float*)d_in[7];
  const float* W1   = (const float*)d_in[8];
  const float* b1   = (const float*)d_in[9];
  const float* g1   = (const float*)d_in[10];
  const float* W2   = (const float*)d_in[11];
  const float* b2   = (const float*)d_in[12];
  const float* g2   = (const float*)d_in[13];
  const float* W3   = (const float*)d_in[14];
  const float* b3   = (const float*)d_in[15];
  const float* Wkv  = (const float*)d_in[16];
  const float* Wout = (const float*)d_in[17];
  float* out = (float*)d_out;

  char* ws = (char*)d_ws;
  size_t off = 0;
  auto alloc = [&](size_t bytes) -> void* {
    void* p = ws + off;
    off = (off + bytes + 255) & ~(size_t)255;
    return p;
  };
  float*  x      = (float*) alloc((size_t)NODES * D   * 4);
  float*  q      = (float*) alloc((size_t)NODES * HID * 4);
  float*  xi     = (float*) alloc((size_t)NODES * D   * 4);
  float*  xj     = (float*) alloc((size_t)NODES * D   * 4);
  float*  h1     = (float*) alloc((size_t)EDGES * RH  * 4);
  float*  h2pre  = (float*) alloc((size_t)EDGES * RH  * 4);
  float*  kvpre  = (float*) alloc((size_t)EDGES * KVD * 4);
  float*  kvbuf  = (float*) alloc((size_t)EDGES * KVD * 4);
  float*  outa   = (float*) alloc((size_t)NODES * HID * 4);
  __bf16* xebf   = (__bf16*)alloc((size_t)EDGES * D   * 2);
  __bf16* h2bf   = (__bf16*)alloc((size_t)EDGES * RH  * 2);
  __bf16* W3bf   = (__bf16*)alloc((size_t)RH * KVD * D * 2);
  __bf16* Wq_t   = (__bf16*)alloc((size_t)HID * D  * 2);
  __bf16* Wxi_t  = (__bf16*)alloc((size_t)D * D    * 2);
  __bf16* Wxj_t  = (__bf16*)alloc((size_t)D * D    * 2);
  __bf16* W2_t   = (__bf16*)alloc((size_t)RH * RH  * 2);
  __bf16* Wkv_t  = (__bf16*)alloc((size_t)KVD * KVD* 2);
  __bf16* Wout_t = (__bf16*)alloc((size_t)D * HID  * 2);

  // ---- weight prep ----
  k_convert_t_bf16<<<(D * HID + 255) / 256, 256, 0, stream>>>(Wq,   Wq_t,   D,   HID);
  k_convert_t_bf16<<<(D * D   + 255) / 256, 256, 0, stream>>>(Wxi,  Wxi_t,  D,   D);
  k_convert_t_bf16<<<(D * D   + 255) / 256, 256, 0, stream>>>(Wxj,  Wxj_t,  D,   D);
  k_convert_t_bf16<<<(RH * RH + 255) / 256, 256, 0, stream>>>(W2,   W2_t,   RH,  RH);
  k_convert_t_bf16<<<(KVD*KVD + 255) / 256, 256, 0, stream>>>(Wkv,  Wkv_t,  KVD, KVD);
  k_convert_t_bf16<<<(HID * D + 255) / 256, 256, 0, stream>>>(Wout, Wout_t, HID, D);
  k_convert_bf16  <<<(RH*KVD*D + 255) / 256, 256, 0, stream>>>(W3, W3bf, RH * KVD * D);

  // ---- node path ----
  k_node_prenorm<<<NODES, 32, 0, stream>>>(features, norm_scale, x);
  k_gemm_bf16_x4<<<(NODES / 16) * (HID / 64), 32, 0, stream>>>(x, Wq_t,  q,  HID, D);
  k_gemm_bf16_x4<<<(NODES / 16) * (D   / 64), 32, 0, stream>>>(x, Wxi_t, xi, D,   D);
  k_gemm_bf16_x4<<<(NODES / 16) * (D   / 64), 32, 0, stream>>>(x, Wxj_t, xj, D,   D);

  // ---- edge path ----
  k_edge_prep<<<EDGES, 32, 0, stream>>>(xi, xj, nidx, rel_dist, W1, b1, g1, xebf, h1);
  k_gemm_bf16_x4<<<(EDGES / 16) * (RH / 64), 32, 0, stream>>>(h1, W2_t, h2pre, RH, RH);
  k_edge_act<<<EDGES, 32, 0, stream>>>(h2pre, b2, g2, h2bf);

  // ---- dominant bilinear GEMM (P @ W3', P built on the fly) ----
  k_kv_gemm<<<EDGES / 16, 128, 0, stream>>>(h2bf, xebf, W3bf, b3, kvpre);

  // ---- kv projection ----
  k_gemm_bf16_x4<<<(EDGES / 16) * (KVD / 64), 32, 0, stream>>>(kvpre, Wkv_t, kvbuf, KVD, KVD);

  // ---- attention + output projection ----
  k_attn<<<NODES * 4, 32, 0, stream>>>(q, kvbuf, nmask, outa);
  k_gemm_bf16_x4<<<(NODES / 16) * (D / 64), 32, 0, stream>>>(outa, Wout_t, out, D, HID);
}
